// ForgetGates_84739704750871
// MI455X (gfx1250) — matrix operand order repaired and checked by reading, not compiled, not run on hardware
//
#include <hip/hip_runtime.h>
#include <hip/hip_bf16.h>

typedef __attribute__((ext_vector_type(16))) _Float16 v16h;
typedef __attribute__((ext_vector_type(8)))  float    v8f;
typedef __attribute__((ext_vector_type(4)))  int      v4i;

#define N_NODES 100000
#define N_EDGES 400000
#define HID     256
#define NT      54
#define NT_PAD  64

// Native no-return f32 global atomic add (STOREcnt-tracked; avoids CAS loop).
__device__ __forceinline__ void fg_atomic_add_f32(float* p, float v) {
    asm volatile("global_atomic_add_f32 %0, %1, off" :: "v"(p), "v"(v) : "memory");
}

// ---------------------------------------------------------------------------
// Kernel 0: zero the output (segment_sum identity; harness poisons d_out).
// ---------------------------------------------------------------------------
__global__ void fg_zero_kernel(float* __restrict__ out, int n) {
    int i = blockIdx.x * blockDim.x + threadIdx.x;
    int stride = gridDim.x * blockDim.x;
    for (; i < n; i += stride) out[i] = 0.0f;
}

// ---------------------------------------------------------------------------
// Kernel 1: prep weights in workspace.
//   Whf16  : W_hf (256x256) -> f16, same layout (B column h == W_hf row h).
//   Wif16p : W_if (256x54)  -> f16 zero-padded to 256x64 (guard-free B frags).
// ---------------------------------------------------------------------------
__global__ void fg_prep_kernel(const float* __restrict__ W_hf,
                               const float* __restrict__ W_if,
                               _Float16* __restrict__ Whf16,
                               _Float16* __restrict__ Wif16p) {
    int i = blockIdx.x * blockDim.x + threadIdx.x;
    int stride = gridDim.x * blockDim.x;
    const int n_total = HID * HID + HID * NT_PAD;
    for (; i < n_total; i += stride) {
        if (i < HID * HID) {
            Whf16[i] = (_Float16)W_hf[i];
        } else {
            const int j = i - HID * HID;
            const int h = j >> 6;
            const int v = j & 63;
            Wif16p[j] = (v < NT) ? (_Float16)W_if[h * NT + v] : (_Float16)0.0f;
        }
    }
}

// ---------------------------------------------------------------------------
// Kernel 2: Wx = x @ W_if^T  -> workspace [N_NODES x 256] f32.
// One wave per 16-node tile. K = 54 padded to 64 -> 2 WMMA k-steps.
// ---------------------------------------------------------------------------
__global__ __launch_bounds__(128) void fg_wx_kernel(const float* __restrict__ x,
                                                    const _Float16* __restrict__ Wif16p,
                                                    float* __restrict__ Wx) {
    const int lane = threadIdx.x & 31;
    const int wave = threadIdx.x >> 5;
    const int tile = blockIdx.x * 4 + wave;           // 16-node tile index
    if (tile >= N_NODES / 16) return;                 // wave-uniform: EXEC stays all-1s
    const int n0   = tile * 16;
    const int sel  = lane >> 4;                       // 0: lanes 0-15, 1: lanes 16-31
    const int half = lane & 15;

    // A fragments (16x32 f16): lane<16 holds K {k0..k0+15} of row M=half,
    // lane>=16 holds K {k0+16..k0+31}. Columns >= 54 are compile-time zeros
    // (select folds, dead loads dropped -> no exec-mask branching).
    v16h a[2];
    const float* xrow = x + (size_t)(n0 + half) * NT;
#pragma unroll
    for (int kk = 0; kk < 2; ++kk) {
        const int k0 = kk * 32;
#pragma unroll
        for (int j = 0; j < 8; ++j) {
            const int c0 = k0 + sel * 16 + j;
            const int c1 = k0 + sel * 16 + 8 + j;
            a[kk][j]     = (c0 < NT) ? (_Float16)xrow[c0] : (_Float16)0.0f;
            a[kk][j + 8] = (c1 < NT) ? (_Float16)xrow[c1] : (_Float16)0.0f;
        }
    }

#pragma unroll 4
    for (int t = 0; t < 16; ++t) {                    // 16 column tiles of 16 -> H=256
        const int h0 = t * 16;
        const _Float16* wrow = Wif16p + (size_t)(h0 + half) * NT_PAD;  // padded B column
        v8f acc = {};
#pragma unroll
        for (int kk = 0; kk < 2; ++kk) {
            v16h b = *(const v16h*)(wrow + kk * 32 + sel * 16);        // 32B aligned
            acc = __builtin_amdgcn_wmma_f32_16x16x32_f16(false, a[kk], false, b,
                                                         (short)0, acc, false, false);
        }
        // D layout: VGPR r -> row r (lanes 0-15) / row r+8 (lanes 16-31), col = h0+half
#pragma unroll
        for (int r = 0; r < 8; ++r) {
            const int row = n0 + r + sel * 8;
            Wx[(size_t)row * HID + h0 + half] = acc[r];
        }
    }
}

// ---------------------------------------------------------------------------
// Kernel 3: main fused kernel. One block (4 waves) per 16-edge tile.
//   Uh = h_tile @ W_hf^T (WMMA f16->f32), then
//   out[seg[e]] += sigmoid(Uh + b_f + Wx[seg[e]]) * c_all[e]
// with run-length pre-reduction over sorted seg before f32 global atomics.
// ---------------------------------------------------------------------------
__global__ __launch_bounds__(128) void fg_main_kernel(const float* __restrict__ h_all,
                                                      const float* __restrict__ c_all,
                                                      const int*   __restrict__ seg,
                                                      const _Float16* __restrict__ Whf16,
                                                      const float* __restrict__ b_f,
                                                      const float* __restrict__ Wx,
                                                      float* __restrict__ out) {
    __shared__ float hs[16 * HID];                    // 16 KB h-tile staging
    const int tid  = threadIdx.x;
    const int lane = tid & 31;
    const int wave = tid >> 5;
    const int e0   = blockIdx.x * 16;

    // Stage the 16x256 f32 h tile into LDS.
#if __has_builtin(__builtin_amdgcn_global_load_async_to_lds_b128)
    {
        // gfx1250 async copy path: 128 threads x 8 x 16B, ASYNCcnt-tracked.
        const float* gsrc = h_all + (size_t)e0 * HID;
#pragma unroll
        for (int it = 0; it < 8; ++it) {
            const int i = (tid + it * 128) * 4;       // float index, 16B granules
            __builtin_amdgcn_global_load_async_to_lds_b128(
                (__attribute__((address_space(1))) v4i*)(gsrc + i),
                (__attribute__((address_space(3))) v4i*)(hs + i),
                0, 0);
        }
#if __has_builtin(__builtin_amdgcn_s_wait_asynccnt)
        __builtin_amdgcn_s_wait_asynccnt(0);
#else
        asm volatile("s_wait_asynccnt 0" ::: "memory");
#endif
    }
#else
    {
        const float4* src = (const float4*)(h_all + (size_t)e0 * HID);
        float4* dst = (float4*)hs;
#pragma unroll
        for (int i = 0; i < 8; ++i) dst[tid + i * 128] = src[tid + i * 128];
    }
#endif
    __syncthreads();

    const int sel  = lane >> 4;
    const int half = lane & 15;

    // Build all 8 A fragments (full K=256) from LDS, f32 -> f16.
    v16h a[8];
    const float* hrow = hs + half * HID;              // A: all lanes hold row M = half
#pragma unroll
    for (int k = 0; k < 8; ++k) {
        const int base = k * 32 + sel * 16;
#pragma unroll
        for (int j = 0; j < 8; ++j) {
            a[k][j]     = (_Float16)hrow[base + j];
            a[k][j + 8] = (_Float16)hrow[base + 8 + j];
        }
    }

    // Segment ids for this lane's 8 output rows (m = sel*8 + r), sorted ascending.
    int segv[8];
#pragma unroll
    for (int r = 0; r < 8; ++r) segv[r] = seg[e0 + sel * 8 + r];

    // Each wave owns 64 columns: h0 = wave*64 + t*16.
    for (int t = 0; t < 4; ++t) {
        const int h0  = wave * 64 + t * 16;
        const int col = h0 + half;
        const _Float16* wcol = Whf16 + (size_t)col * HID;  // contiguous B column
        v8f acc = {};
#pragma unroll
        for (int k = 0; k < 8; ++k) {
            // B: lane<16 -> K [k*32, k*32+16), lane>=16 -> K [k*32+16, k*32+32); 32B aligned
            v16h b = *(const v16h*)(wcol + k * 32 + sel * 16);
            acc = __builtin_amdgcn_wmma_f32_16x16x32_f16(false, a[k], false, b,
                                                         (short)0, acc, false, false);
        }

        const float bias = b_f[col];
        int   cur    = segv[0];
        float runsum = 0.0f;
#pragma unroll
        for (int r = 0; r < 8; ++r) {
            const int m = sel * 8 + r;                 // local edge row of acc[r]
            const int s = segv[r];
            const float z = acc[r] + bias + Wx[(size_t)s * HID + col];
            const float f = 1.0f / (1.0f + __expf(-z));
            const float contrib = f * c_all[(size_t)(e0 + m) * HID + col];
            if (s != cur) {                            // sorted seg: flush finished run
                fg_atomic_add_f32(out + (size_t)cur * HID + col, runsum);
                cur = s;
                runsum = 0.0f;
            }
            runsum += contrib;
        }
        fg_atomic_add_f32(out + (size_t)cur * HID + col, runsum);
    }
}

// ---------------------------------------------------------------------------
extern "C" void kernel_launch(void* const* d_in, const int* in_sizes, int n_in,
                              void* d_out, int out_size, void* d_ws, size_t ws_size,
                              hipStream_t stream) {
    const float* x     = (const float*)d_in[0];
    const float* h_all = (const float*)d_in[1];
    const float* c_all = (const float*)d_in[2];
    const int*   seg   = (const int*)  d_in[3];
    const float* W_if  = (const float*)d_in[4];
    const float* W_hf  = (const float*)d_in[5];
    const float* b_f   = (const float*)d_in[6];
    float* out = (float*)d_out;

    // Workspace layout:
    //   [Wx : N_NODES*HID f32][Whf16 : HID*HID f16][Wif16p : HID*64 f16]
    char* ws = (char*)d_ws;
    float*    Wx     = (float*)ws;
    _Float16* Whf16  = (_Float16*)(ws + (size_t)N_NODES * HID * sizeof(float));
    _Float16* Wif16p = (_Float16*)(ws + (size_t)N_NODES * HID * sizeof(float)
                                      + (size_t)HID * HID * sizeof(_Float16));

    // 0) zero output
    fg_zero_kernel<<<2048, 256, 0, stream>>>(out, N_NODES * HID);
    // 1) weights -> f16 (W_if zero-padded to K=64)
    fg_prep_kernel<<<320, 256, 0, stream>>>(W_hf, W_if, Whf16, Wif16p);
    // 2) Wx = x @ W_if^T   (6250 node tiles, 4 waves/block)
    fg_wx_kernel<<<(N_NODES / 16 + 3) / 4, 128, 0, stream>>>(x, Wif16p, Wx);
    // 3) fused Uh GEMM + gate + segment-sum (25000 edge tiles)
    fg_main_kernel<<<N_EDGES / 16, 128, 0, stream>>>(h_all, c_all, seg, Whf16, b_f, Wx, out);
}
